// MultiHeadedAttention_9715216024037
// MI455X (gfx1250) — matrix operand order repaired and checked by reading, not compiled, not run on hardware
//
#include <hip/hip_runtime.h>

// MultiHeadedAttention forward for B=2, S=2048, D=1024, H=16, DK=64 on gfx1250.
// All matrix math via v_wmma_f32_16x16x32_f16 (wave32 WMMA), f32 accumulate.
// Round 4: softmax row-sums computed by WMMA (P @ ones accumulator) instead of
// DPP butterflies; mask folded into an additive bias (fma instead of mul+sel).

#define BB   2
#define SS   2048
#define DD   1024
#define HH   16
#define DK   64
#define NROWS (BB * SS)          // 4096

typedef __attribute__((ext_vector_type(16))) _Float16 v16h;
typedef __attribute__((ext_vector_type(8)))  _Float16 v8h;
typedef __attribute__((ext_vector_type(8)))  float    v8f;
typedef __attribute__((ext_vector_type(4)))  float    v4f;

union AFrag { v16h v; v8h u[2]; _Float16 h[16]; };

__device__ __forceinline__ int   f2i(float x) { union { float f; int i; } u; u.f = x; return u.i; }
__device__ __forceinline__ float i2f(int x)   { union { float f; int i; } u; u.i = x; return u.f; }

// 16-lane max butterfly via DPP16 (lanes 0-15 and 16-31 independently):
// quad_perm xor1 (0xB1), quad_perm xor2 (0x4E), row_half_mirror, row_mirror.
__device__ __forceinline__ float dpp_max16(float v) {
    v = fmaxf(v, i2f(__builtin_amdgcn_update_dpp(0, f2i(v), 0xB1,  0xF, 0xF, true)));
    v = fmaxf(v, i2f(__builtin_amdgcn_update_dpp(0, f2i(v), 0x4E,  0xF, 0xF, true)));
    v = fmaxf(v, i2f(__builtin_amdgcn_update_dpp(0, f2i(v), 0x141, 0xF, 0xF, true)));
    v = fmaxf(v, i2f(__builtin_amdgcn_update_dpp(0, f2i(v), 0x140, 0xF, 0xF, true)));
    return v;
}

// ---------------------------------------------------------------------------
// Elementwise f32 -> f16 pack (vectorized 8 elements/thread).
// ---------------------------------------------------------------------------
__global__ __launch_bounds__(256)
void pack_f16_kernel(const float* __restrict__ X, _Float16* __restrict__ Y, int n8)
{
    const int i = blockIdx.x * blockDim.x + threadIdx.x;
    if (i < n8) {
        const v4f a = *reinterpret_cast<const v4f*>(X + (size_t)i * 8);
        const v4f b = *reinterpret_cast<const v4f*>(X + (size_t)i * 8 + 4);
        v8h o;
        o[0] = (_Float16)a.x; o[1] = (_Float16)a.y;
        o[2] = (_Float16)a.z; o[3] = (_Float16)a.w;
        o[4] = (_Float16)b.x; o[5] = (_Float16)b.y;
        o[6] = (_Float16)b.z; o[7] = (_Float16)b.w;
        *reinterpret_cast<v8h*>(Y + (size_t)i * 8) = o;
    }
}

// ---------------------------------------------------------------------------
// GEMM: Y = A(f16, MxK row-major) @ W^T(f16, NxK row-major) + bias.
// One wave per block computes a 32(M) x 64(N) tile; K-loop in chunks of 32.
// OUT = 0: f16 row-major   OUT = 1: f16 (B,H,DK,S) transposed (for V)
// OUT = 2: f32 row-major (final output projection)
// ---------------------------------------------------------------------------
template <int OUT>
__global__ __launch_bounds__(32)
void gemm_f16_kernel(const _Float16* __restrict__ A, const _Float16* __restrict__ W,
                     const float* __restrict__ bias, void* __restrict__ Yv)
{
    const int lane = threadIdx.x & 31;
    const int g    = lane >> 4;      // lane half (0: lanes 0-15, 1: lanes 16-31)
    const int ln   = lane & 15;
    const int m0   = blockIdx.y * 32;
    const int n0   = blockIdx.x * 64;

    v8f acc[2][4] = {};

    for (int kb = 0; kb < DD; kb += 32) {
        v16h af[2];
#pragma unroll
        for (int mh = 0; mh < 2; ++mh) {
            const _Float16* arow = A + (size_t)(m0 + 16 * mh + ln) * DD + kb;
            AFrag Af;
            Af.u[0] = *reinterpret_cast<const v8h*>(arow + 8 * g);
            Af.u[1] = *reinterpret_cast<const v8h*>(arow + 16 + 8 * g);
            af[mh] = Af.v;
        }
#pragma unroll
        for (int t = 0; t < 4; ++t) {
            // B fragment: col n = n0+16t+ln (row of W), K at 8g..8g+7 and 16+8g..16+8g+7
            const _Float16* brow = W + (size_t)(n0 + 16 * t + ln) * DD + kb + 8 * g;
            AFrag Bf;
            Bf.u[0] = *reinterpret_cast<const v8h*>(brow);
            Bf.u[1] = *reinterpret_cast<const v8h*>(brow + 16);
#pragma unroll
            for (int mh = 0; mh < 2; ++mh)
                acc[mh][t] = __builtin_amdgcn_wmma_f32_16x16x32_f16(
                    false, af[mh], false, Bf.v, (short)0, acc[mh][t], false, false);
        }
    }

#pragma unroll
    for (int t = 0; t < 4; ++t) {
        const int   col = n0 + 16 * t + ln;
        const float bv  = bias[col];
#pragma unroll
        for (int mh = 0; mh < 2; ++mh) {
#pragma unroll
            for (int r = 0; r < 8; ++r) {
                const int   row = m0 + 16 * mh + r + 8 * g; // C/D: M = r + 8*(lane>=16)
                const float val = acc[mh][t][r] + bv;
                if (OUT == 0) {
                    ((_Float16*)Yv)[(size_t)row * DD + col] = (_Float16)val;
                } else if (OUT == 1) {
                    const int b  = row >> 11, s  = row & (SS - 1);
                    const int hh = col >> 6,  dk = col & (DK - 1);
                    ((_Float16*)Yv)[((size_t)(b * HH + hh) * DK + dk) * SS + s] =
                        (_Float16)val;
                } else {
                    ((float*)Yv)[(size_t)row * DD + col] = val;
                }
            }
        }
    }
}

// ---------------------------------------------------------------------------
// Flash attention: one wave per (batch, head, 32-query block). Streams 32-key
// blocks: 8 score WMMAs (K fragments shared by both query halves) -> online
// softmax (DPP max only; row sums via P @ ones WMMA accumulator) -> P
// transposed via LDS into A-fragment layout -> 10 accumulate WMMAs.
// ---------------------------------------------------------------------------
__global__ __launch_bounds__(32)
void flash_attn_kernel(const _Float16* __restrict__ Q,
                       const _Float16* __restrict__ K,
                       const _Float16* __restrict__ Vt,   // (B,H,DK,S)
                       const int* __restrict__ mask,      // (B,1,1,S)
                       _Float16* __restrict__ Ctx)        // (B*S, D)
{
    __shared__ __align__(32) _Float16 lds_p[2][16 * 32];  // P tiles: 2 x (16q x 32k)

    const int lane = threadIdx.x & 31;
    const int g    = lane >> 4;
    const int ln   = lane & 15;
    const int b    = blockIdx.z;
    const int h    = blockIdx.y;
    const int q0   = blockIdx.x * 32;

    // Q A-fragments: 2 query halves x 2 K-chunks (DK=64), loaded once
    v16h aq[2][2];
#pragma unroll
    for (int mh = 0; mh < 2; ++mh) {
        const _Float16* Qbase =
            Q + (size_t)(b * SS + q0 + 16 * mh + ln) * DD + h * DK;
#pragma unroll
        for (int kb = 0; kb < 2; ++kb) {
            AFrag A;
            A.u[0] = *reinterpret_cast<const v8h*>(Qbase + 32 * kb + 16 * g);
            A.u[1] = *reinterpret_cast<const v8h*>(Qbase + 32 * kb + 16 * g + 8);
            aq[mh][kb] = A.v;
        }
    }

    // All-ones B fragment: every column of P @ ones is the row-sum of P.
    v16h bones;
#pragma unroll
    for (int i = 0; i < 16; ++i) bones[i] = (_Float16)1.0f;

    float mrow[2][8];
    v8f   co[2][4] = {};
    v8f   lco[2]   = {};                 // running softmax denominators (row sums)
#pragma unroll
    for (int mh = 0; mh < 2; ++mh)
#pragma unroll
        for (int r = 0; r < 8; ++r) mrow[mh][r] = -3.0e38f;

    const int* mline = mask + b * SS;

    for (int n0 = 0; n0 < SS; n0 += 32) {
        // ---- scores: K fragments loaded once, used by both query halves ----
        v8f sc[2][2];                                     // [query half][key tile]
#pragma unroll
        for (int t = 0; t < 2; ++t) {
            const int key = n0 + 16 * t + ln;
            const _Float16* Kbase = K + (size_t)(b * SS + key) * DD + h * DK;
            if (n0 + 32 < SS)
                __builtin_prefetch(Kbase + 32 * DD, 0, 1);  // global_prefetch_b8
            const v16h bk0 = *reinterpret_cast<const v16h*>(Kbase + 16 * g);
            const v16h bk1 = *reinterpret_cast<const v16h*>(Kbase + 32 + 16 * g);
#pragma unroll
            for (int mh = 0; mh < 2; ++mh) {
                v8f c = {};
                c = __builtin_amdgcn_wmma_f32_16x16x32_f16(
                    false, aq[mh][0], false, bk0, (short)0, c, false, false);
                c = __builtin_amdgcn_wmma_f32_16x16x32_f16(
                    false, aq[mh][1], false, bk1, (short)0, c, false, false);
                sc[mh][t] = c;
            }
        }

        // ---- scale + additive key-mask bias: s' = s*0.125 + {0 | -1e9} ----
#pragma unroll
        for (int t = 0; t < 2; ++t) {
            const float mb = (mline[n0 + 16 * t + ln] == 0) ? -1.0e9f : 0.0f;
#pragma unroll
            for (int mh = 0; mh < 2; ++mh)
#pragma unroll
                for (int r = 0; r < 8; ++r)
                    sc[mh][t][r] = fmaf(sc[mh][t][r], 0.125f, mb);
        }

        // ---- online softmax (DPP max), rescale, stage P in LDS ----
#pragma unroll
        for (int mh = 0; mh < 2; ++mh) {
#pragma unroll
            for (int r = 0; r < 8; ++r) {
                const float rmax  = dpp_max16(fmaxf(sc[mh][0][r], sc[mh][1][r]));
                const float newm  = fmaxf(mrow[mh][r], rmax);
                const float alpha = __expf(mrow[mh][r] - newm);
                const float p0    = __expf(sc[mh][0][r] - newm);
                const float p1    = __expf(sc[mh][1][r] - newm);
                mrow[mh][r] = newm;
                lco[mh][r] *= alpha;
#pragma unroll
                for (int t = 0; t < 4; ++t) co[mh][t][r] *= alpha;

                const int m = r + 8 * g;                  // row in C/D layout
                lds_p[mh][m * 32 + ln]      = (_Float16)p0;
                lds_p[mh][m * 32 + 16 + ln] = (_Float16)p1;
            }
        }
        __syncthreads();   // single-wave WG: barrier->nop, DS ops stay in-order

        // ---- reload P as A-fragments (16 x 32 each) ----
        AFrag Ap[2];
#pragma unroll
        for (int mh = 0; mh < 2; ++mh) {
            Ap[mh].u[0] = *reinterpret_cast<const v8h*>(&lds_p[mh][ln * 32 + 16 * g]);
            Ap[mh].u[1] = *reinterpret_cast<const v8h*>(&lds_p[mh][ln * 32 + 16 * g + 8]);
        }

        // ---- row sums on the matrix pipe: lco = lco*alpha + P @ ones ----
#pragma unroll
        for (int mh = 0; mh < 2; ++mh)
            lco[mh] = __builtin_amdgcn_wmma_f32_16x16x32_f16(
                false, Ap[mh].v, false, bones, (short)0, lco[mh], false, false);

        // ---- context: V fragments loaded once, used by both query halves ----
#pragma unroll
        for (int t = 0; t < 4; ++t) {
            const int dk = 16 * t + ln;
            const _Float16* Vrow =
                Vt + ((size_t)(b * HH + h) * DK + dk) * SS + n0 + 16 * g;
            if (n0 + 32 < SS) __builtin_prefetch(Vrow + 32, 0, 1);
            const v16h bf = *reinterpret_cast<const v16h*>(Vrow);
#pragma unroll
            for (int mh = 0; mh < 2; ++mh)
                co[mh][t] = __builtin_amdgcn_wmma_f32_16x16x32_f16(
                    false, Ap[mh].v, false, bf, (short)0, co[mh][t], false, false);
        }
        __syncthreads();
    }

    // ---- normalize + store context tiles (B*S, D) ----
#pragma unroll
    for (int mh = 0; mh < 2; ++mh) {
#pragma unroll
        for (int t = 0; t < 4; ++t) {
            const int col = h * DK + 16 * t + ln;
#pragma unroll
            for (int r = 0; r < 8; ++r) {
                const int   row  = q0 + 16 * mh + r + 8 * g;
                const float invl = 1.0f / lco[mh][r];     // row sum (any lane)
                Ctx[(size_t)(b * SS + row) * DD + col] =
                    (_Float16)(co[mh][t][r] * invl);
            }
        }
    }
}

// ---------------------------------------------------------------------------
extern "C" void kernel_launch(void* const* d_in, const int* in_sizes, int n_in,
                              void* d_out, int out_size, void* d_ws, size_t ws_size,
                              hipStream_t stream)
{
    (void)in_sizes; (void)n_in; (void)out_size; (void)ws_size;

    const float* query = (const float*)d_in[0];
    const float* key   = (const float*)d_in[1];
    const float* value = (const float*)d_in[2];
    const int*   mask  = (const int*)  d_in[3];
    const float* Wq    = (const float*)d_in[4];
    const float* bq    = (const float*)d_in[5];
    const float* Wk    = (const float*)d_in[6];
    const float* bk    = (const float*)d_in[7];
    const float* Wv    = (const float*)d_in[8];
    const float* bv    = (const float*)d_in[9];
    const float* Wo    = (const float*)d_in[10];
    const float* bo    = (const float*)d_in[11];
    float*       out   = (float*)d_out;

    // Workspace layout (f16 tensors):
    //   Qh, Kh, Vth, Ctx : 4096x1024 each  (8 MiB each, 32 MiB)
    //   Qx, Kx, Vx       : packed activations (8 MiB each, 24 MiB)
    //   Wqh,Wkh,Wvh,Woh  : packed weights (2 MiB each, 8 MiB)   total 64 MiB
    const size_t act_bytes = (size_t)NROWS * DD * sizeof(_Float16);  // 8 MiB
    const size_t w_bytes   = (size_t)DD * DD * sizeof(_Float16);     // 2 MiB
    char* ws = (char*)d_ws;
    _Float16* Qh  = (_Float16*)(ws + 0 * act_bytes);
    _Float16* Kh  = (_Float16*)(ws + 1 * act_bytes);
    _Float16* Vth = (_Float16*)(ws + 2 * act_bytes);
    _Float16* Ctx = (_Float16*)(ws + 3 * act_bytes);
    _Float16* Qx  = (_Float16*)(ws + 4 * act_bytes);
    _Float16* Kx  = (_Float16*)(ws + 5 * act_bytes);
    _Float16* Vx  = (_Float16*)(ws + 6 * act_bytes);
    char*     wsw = ws + 7 * act_bytes;
    _Float16* Wqh = (_Float16*)(wsw + 0 * w_bytes);
    _Float16* Wkh = (_Float16*)(wsw + 1 * w_bytes);
    _Float16* Wvh = (_Float16*)(wsw + 2 * w_bytes);
    _Float16* Woh = (_Float16*)(wsw + 3 * w_bytes);

    // ---- pack f32 -> f16 ----
    const int act_n8 = NROWS * DD / 8;   // 524288
    const int w_n8   = DD * DD / 8;      // 131072
    pack_f16_kernel<<<(act_n8 + 255) / 256, 256, 0, stream>>>(query, Qx, act_n8);
    pack_f16_kernel<<<(act_n8 + 255) / 256, 256, 0, stream>>>(key,   Kx, act_n8);
    pack_f16_kernel<<<(act_n8 + 255) / 256, 256, 0, stream>>>(value, Vx, act_n8);
    pack_f16_kernel<<<(w_n8 + 255) / 256, 256, 0, stream>>>(Wq, Wqh, w_n8);
    pack_f16_kernel<<<(w_n8 + 255) / 256, 256, 0, stream>>>(Wk, Wkh, w_n8);
    pack_f16_kernel<<<(w_n8 + 255) / 256, 256, 0, stream>>>(Wv, Wvh, w_n8);
    pack_f16_kernel<<<(w_n8 + 255) / 256, 256, 0, stream>>>(Wo, Woh, w_n8);

    // ---- projections (pure-f16 WMMA GEMMs) ----
    const dim3 gemm_grid(DD / 64, NROWS / 32);      // (16, 128)
    gemm_f16_kernel<0><<<gemm_grid, 32, 0, stream>>>(Qx, Wqh, bq, (void*)Qh);
    gemm_f16_kernel<0><<<gemm_grid, 32, 0, stream>>>(Kx, Wkh, bk, (void*)Kh);
    gemm_f16_kernel<1><<<gemm_grid, 32, 0, stream>>>(Vx, Wvh, bv, (void*)Vth);

    // ---- attention (32 queries per wave) ----
    const dim3 attn_grid(SS / 32, HH, BB);          // (64, 16, 2)
    flash_attn_kernel<<<attn_grid, 32, 0, stream>>>(Qh, Kh, Vth, mask, Ctx);

    // ---- output projection (f32 out + bias) ----
    gemm_f16_kernel<2><<<gemm_grid, 32, 0, stream>>>(Ctx, Woh, bo, (void*)out);
}
